// EGNN_dynamics_transferable_MD_11330123727053
// MI455X (gfx1250) — matrix operand order repaired and checked by reading, compiled, not run on hardware
//
#include <hip/hip_runtime.h>
#include <hip/hip_bf16.h>

typedef _Float16 h16;
typedef __attribute__((ext_vector_type(16))) _Float16 v16h;
typedef __attribute__((ext_vector_type(8)))  _Float16 v8h;
typedef __attribute__((ext_vector_type(8)))  float    v8f;

#define NBATCH 64
#define NNODE  128
#define HDIM   64
#define BNODES (NBATCH*NNODE)
#define NLAYER 4
#define KEDGE  160   // 2H+2=130 padded to 5 K-tiles of 32

__device__ __forceinline__ float silu_f(float x) { return x / (1.0f + __expf(-x)); }

__device__ __forceinline__ v8f wmma_f16(v16h a, v16h b, v8f c) {
  // D = A(16x32 f16) x B(32x16 f16) + C(16x16 f32)
  return __builtin_amdgcn_wmma_f32_16x16x32_f16(false, a, false, b, (short)0, c, false, false);
}

__device__ __forceinline__ v16h pack16(v8h lo, v8h hi) {
  v16h r;
#pragma unroll
  for (int q = 0; q < 8; ++q) { r[q] = lo[q]; r[8 + q] = hi[q]; }
  return r;
}

__device__ __forceinline__ v8f zero8() {
  v8f z;
#pragma unroll
  for (int r = 0; r < 8; ++r) z[r] = 0.f;
  return z;
}

// A fragment (16x32 f16) from row-major f16 LDS tile.
// ISA layout: lane L -> M=L&15; halves 0..7 = K kb..kb+7, halves 8..15 = K kb+16..kb+23,
// kb = k0 + ((L>=16) ? 8 : 0).
__device__ __forceinline__ v16h load_a_rm(const h16* base, int stride, int lane, int k0) {
  int M  = lane & 15;
  int kb = k0 + ((lane & 16) ? 8 : 0);
  const h16* p = base + M * stride + kb;
  return pack16(*(const v8h*)p, *(const v8h*)(p + 16));
}

// A fragment where all 16 M-rows replicate one source row (h_i broadcast).
__device__ __forceinline__ v16h load_a_bcast(const h16* row, int lane, int k0) {
  int kb = k0 + ((lane & 16) ? 8 : 0);
  const h16* p = row + kb;
  return pack16(*(const v8h*)p, *(const v8h*)(p + 16));
}

// B fragment (32x16 f16) from weights stored transposed [N][Kpad]:
// lane L -> N = n0+(L&15); halves 0..7 = K kb..kb+7, halves 8..15 = K kb+16..kb+23, kb = k0 + ((L>=16) ? 8 : 0).
__device__ __forceinline__ v16h load_b_tr(const h16* w, int kpad, int lane, int k0, int n0) {
  int N  = n0 + (lane & 15);
  int kb = k0 + ((lane & 16) ? 8 : 0);
  const h16* p = w + N * kpad + kb;
  return pack16(*(const v8h*)p, *(const v8h*)(p + 16));
}

// C/D fragment (16x16 f32) -> row-major f16 LDS tile.
// lane L -> N = n0+(L&15), rows r + ((L>=16) ? 8 : 0).
__device__ __forceinline__ void store_c_rm(h16* dst, int stride, int lane, int n0, v8f c) {
  int N  = n0 + (lane & 15);
  int Mb = (lane & 16) ? 8 : 0;
#pragma unroll
  for (int r = 0; r < 8; ++r) dst[(Mb + r) * stride + N] = (h16)c[r];
}

// ---------------------------------------------------------------------------
// Prep: convert weights to f16, transposed [N][Kpad] (zero-padded K).
// ---------------------------------------------------------------------------
__global__ void egnn_prep_weights(
    const float* __restrict__ We1, const float* __restrict__ We2,
    const float* __restrict__ Wc1, const float* __restrict__ Wc2,
    const float* __restrict__ Wn1, const float* __restrict__ Wn2,
    h16* __restrict__ We1h, h16* __restrict__ We2h, h16* __restrict__ Wc1h,
    h16* __restrict__ Wc2h, h16* __restrict__ Wn1h, h16* __restrict__ Wn2h) {
  int idx = blockIdx.x * blockDim.x + threadIdx.x;
  const int SZ1 = NLAYER * HDIM * KEDGE;   // We1h
  const int SZ2 = NLAYER * HDIM * HDIM;    // We2h / Wc1h / Wn2h
  const int SZ4 = NLAYER * HDIM;           // Wc2h
  const int SZ5 = NLAYER * HDIM * 128;     // Wn1h
  if (idx < SZ1) {
    int l = idx / (HDIM * KEDGE), r = idx % (HDIM * KEDGE);
    int n = r / KEDGE, k = r % KEDGE;
    We1h[idx] = (h16)((k < 130) ? We1[(l * 130 + k) * HDIM + n] : 0.f);
    return;
  }
  idx -= SZ1;
  if (idx < SZ2) {
    int l = idx / (HDIM * HDIM), r = idx % (HDIM * HDIM);
    int n = r / HDIM, k = r % HDIM;
    We2h[idx] = (h16)We2[(l * HDIM + k) * HDIM + n];
    return;
  }
  idx -= SZ2;
  if (idx < SZ2) {
    int l = idx / (HDIM * HDIM), r = idx % (HDIM * HDIM);
    int n = r / HDIM, k = r % HDIM;
    Wc1h[idx] = (h16)Wc1[(l * HDIM + k) * HDIM + n];
    return;
  }
  idx -= SZ2;
  if (idx < SZ4) { Wc2h[idx] = (h16)Wc2[idx]; return; }
  idx -= SZ4;
  if (idx < SZ5) {
    int l = idx / (HDIM * 128), r = idx % (HDIM * 128);
    int n = r / 128, k = r % 128;
    Wn1h[idx] = (h16)Wn1[(l * 128 + k) * HDIM + n];
    return;
  }
  idx -= SZ5;
  if (idx < SZ2) {
    int l = idx / (HDIM * HDIM), r = idx % (HDIM * HDIM);
    int n = r / HDIM, k = r % HDIM;
    Wn2h[idx] = (h16)Wn2[(l * HDIM + k) * HDIM + n];
  }
}

// ---------------------------------------------------------------------------
// Prep: masked x/x0, node embedding  h0 = [h*nm, t*nm] @ Wemb + bemb (K=17).
// ---------------------------------------------------------------------------
__global__ void egnn_prep_nodes(
    const float* __restrict__ t, const float* __restrict__ xs,
    const float* __restrict__ hfeat, const float* __restrict__ nmask,
    const float* __restrict__ Wemb, const float* __restrict__ bemb,
    float* __restrict__ x0, float* __restrict__ xA, float* __restrict__ hA) {
  int tid = blockIdx.x * blockDim.x + threadIdx.x;
  if (tid < BNODES * HDIM) {
    int node = tid >> 6, o = tid & 63;
    float nm = nmask[node];
    float acc = bemb[o];
#pragma unroll
    for (int k = 0; k < 16; ++k) acc += hfeat[node * 16 + k] * nm * Wemb[k * HDIM + o];
    acc += t[node >> 7] * nm * Wemb[16 * HDIM + o];
    hA[tid] = acc;
  }
  if (tid < BNODES * 3) {
    float v = xs[tid] * nmask[tid / 3];
    x0[tid] = v; xA[tid] = v;
  }
}

// ---------------------------------------------------------------------------
// One fused E_GCL layer. Block = (batch b, 16 rows i). 256 threads = 8 waves.
// Each wave owns 2 rows i and all 128 edges (i, j) per row, fully fused:
// edge MLP (2 GEMMs) -> coord MLP (GEMM + dot) -> local segment sums ->
// node MLP (2 GEMMs) -> residual h update + x update.
// ---------------------------------------------------------------------------
__global__ __launch_bounds__(256)
void egnn_layer(
    const float* __restrict__ xIn, float* __restrict__ xOut,
    const float* __restrict__ hIn, float* __restrict__ hOut,
    const float* __restrict__ x0g,
    const float* __restrict__ nmask, const float* __restrict__ emask,
    const h16* __restrict__ We1h, const h16* __restrict__ We2h,
    const h16* __restrict__ Wc1h, const h16* __restrict__ Wc2h,
    const h16* __restrict__ Wn1h, const h16* __restrict__ Wn2h,
    const float* __restrict__ be1, const float* __restrict__ be2,
    const float* __restrict__ bc1, const float* __restrict__ bn1,
    const float* __restrict__ bn2) {
  __shared__ h16   sH[NNODE * HDIM];      // hcur, whole batch, f16
  __shared__ float sX[NNODE][4];
  __shared__ float sX0[NNODE][4];
  __shared__ h16   sWe1[HDIM * KEDGE];    // transposed [N][Kpad]
  __shared__ h16   sWe2[HDIM * HDIM];
  __shared__ h16   sWc1[HDIM * HDIM];
  __shared__ h16   sWc2[HDIM];
  __shared__ h16   sWn1[HDIM * 128];
  __shared__ h16   sWn2[HDIM * HDIM];
  __shared__ float sB[5][HDIM];           // be1,be2,bc1,bn1,bn2
  __shared__ h16   sTmp[8][16 * HDIM];    // per-wave GEMM handoff tile
  __shared__ float sXacc[16][4];          // coord scatter-sum per local row
  __shared__ h16   sNode[16][128];        // [hcur | magg] for node MLP

  const int b    = blockIdx.y;
  const int r0   = blockIdx.x * 16;
  const int tid  = threadIdx.x;
  const int wv   = tid >> 5;
  const int lane = tid & 31;
  const int nb   = b * NNODE;

  for (int idx = tid; idx < NNODE * HDIM; idx += 256) sH[idx] = (h16)hIn[nb * HDIM + idx];
  for (int idx = tid; idx < NNODE; idx += 256) {
    int g = nb + idx;
    sX[idx][0]  = xIn[g * 3 + 0]; sX[idx][1]  = xIn[g * 3 + 1];
    sX[idx][2]  = xIn[g * 3 + 2]; sX[idx][3]  = 0.f;
    sX0[idx][0] = x0g[g * 3 + 0]; sX0[idx][1] = x0g[g * 3 + 1];
    sX0[idx][2] = x0g[g * 3 + 2]; sX0[idx][3] = 0.f;
  }
  for (int idx = tid; idx < HDIM * KEDGE; idx += 256) sWe1[idx] = We1h[idx];
  for (int idx = tid; idx < HDIM * HDIM; idx += 256) {
    sWe2[idx] = We2h[idx]; sWc1[idx] = Wc1h[idx]; sWn2[idx] = Wn2h[idx];
  }
  for (int idx = tid; idx < HDIM * 128; idx += 256) sWn1[idx] = Wn1h[idx];
  if (tid < HDIM) {
    sWc2[tid]  = Wc2h[tid];
    sB[0][tid] = be1[tid]; sB[1][tid] = be2[tid]; sB[2][tid] = bc1[tid];
    sB[3][tid] = bn1[tid]; sB[4][tid] = bn2[tid];
  }
  if (tid < 16) { sXacc[tid][0] = 0.f; sXacc[tid][1] = 0.f; sXacc[tid][2] = 0.f; sXacc[tid][3] = 0.f; }
  for (int idx = tid; idx < 16 * HDIM; idx += 256) {
    int ri = idx >> 6, ch = idx & 63;
    sNode[ri][ch] = (h16)hIn[(nb + r0 + ri) * HDIM + ch];
  }
  __syncthreads();

  h16* myT = sTmp[wv];
  const int Nl = lane & 15;
  const int Mb = (lane & 16) ? 8 : 0;

  for (int rr = 0; rr < 2; ++rr) {
    const int ri = wv * 2 + rr;     // local row
    const int i  = r0 + ri;         // node-in-batch
    const float xi0 = sX[i][0], xi1 = sX[i][1], xi2 = sX[i][2];
    const float y0  = sX0[i][0], y1 = sX0[i][1], y2 = sX0[i][2];
    const float* emrow = emask + (b * NNODE + i) * NNODE;
    float magg[4] = {0.f, 0.f, 0.f, 0.f};

    for (int jt = 0; jt < 8; ++jt) {
      const int j0 = jt * 16;
      // geometry for this lane's edge (i, j0 + (lane&15))
      const int jme = j0 + Nl;
      const float dx0 = xi0 - sX[jme][0], dx1 = xi1 - sX[jme][1], dx2 = xi2 - sX[jme][2];
      const float radial = dx0 * dx0 + dx1 * dx1 + dx2 * dx2;
      const float e0 = y0 - sX0[jme][0], e1 = y1 - sX0[jme][1], e2 = y2 - sX0[jme][2];
      const float eattr = e0 * e0 + e1 * e1 + e2 * e2;

      // ---- edge MLP 1: ef[16,160] @ We1[160,64] ; ef assembled as fragments
      v8f acc[4] = {zero8(), zero8(), zero8(), zero8()};
      {
        v16h a = load_a_bcast(&sH[i * HDIM], lane, 0);            // K 0..31  = h_i
#pragma unroll
        for (int nt = 0; nt < 4; ++nt) acc[nt] = wmma_f16(a, load_b_tr(sWe1, KEDGE, lane, 0, nt * 16), acc[nt]);
        a = load_a_bcast(&sH[i * HDIM], lane, 32);                // K 32..63 = h_i
#pragma unroll
        for (int nt = 0; nt < 4; ++nt) acc[nt] = wmma_f16(a, load_b_tr(sWe1, KEDGE, lane, 32, nt * 16), acc[nt]);
        a = load_a_rm(&sH[j0 * HDIM], HDIM, lane, 0);             // K 64..95 = h_j
#pragma unroll
        for (int nt = 0; nt < 4; ++nt) acc[nt] = wmma_f16(a, load_b_tr(sWe1, KEDGE, lane, 64, nt * 16), acc[nt]);
        a = load_a_rm(&sH[j0 * HDIM], HDIM, lane, 32);            // K 96..127 = h_j
#pragma unroll
        for (int nt = 0; nt < 4; ++nt) acc[nt] = wmma_f16(a, load_b_tr(sWe1, KEDGE, lane, 96, nt * 16), acc[nt]);
        v16h a4;                                                   // K 128..159 = radial, eattr, 0-pad
#pragma unroll
        for (int q = 0; q < 16; ++q) a4[q] = (h16)0.f;
        if (!(lane & 16)) { a4[0] = (h16)radial; a4[1] = (h16)eattr; }
#pragma unroll
        for (int nt = 0; nt < 4; ++nt) acc[nt] = wmma_f16(a4, load_b_tr(sWe1, KEDGE, lane, 128, nt * 16), acc[nt]);
      }
#pragma unroll
      for (int nt = 0; nt < 4; ++nt) {
        float bia = sB[0][nt * 16 + Nl];
#pragma unroll
        for (int r = 0; r < 8; ++r) acc[nt][r] = silu_f(acc[nt][r] + bia);
        store_c_rm(myT, HDIM, lane, nt * 16, acc[nt]);
      }

      // ---- edge MLP 2 + mask: m = silu(t1 @ We2 + be2) * em ; magg partials
      v8f m[4] = {zero8(), zero8(), zero8(), zero8()};
#pragma unroll
      for (int kt = 0; kt < 2; ++kt) {
        v16h a = load_a_rm(myT, HDIM, lane, kt * 32);
#pragma unroll
        for (int nt = 0; nt < 4; ++nt) m[nt] = wmma_f16(a, load_b_tr(sWe2, HDIM, lane, kt * 32, nt * 16), m[nt]);
      }
      float em8[8];
#pragma unroll
      for (int r = 0; r < 8; ++r) em8[r] = emrow[j0 + Mb + r];
#pragma unroll
      for (int nt = 0; nt < 4; ++nt) {
        float bia = sB[1][nt * 16 + Nl];
        float s = 0.f;
#pragma unroll
        for (int r = 0; r < 8; ++r) { m[nt][r] = silu_f(m[nt][r] + bia) * em8[r]; s += m[nt][r]; }
        s += __shfl_xor(s, 16, 32);   // combine the two M-halves per channel
        magg[nt] += s;
        store_c_rm(myT, HDIM, lane, nt * 16, m[nt]);
      }

      // ---- coord MLP: cp = silu(m @ Wc1 + bc1); c = cp @ Wc2 (H->1)
      v8f cp[4] = {zero8(), zero8(), zero8(), zero8()};
#pragma unroll
      for (int kt = 0; kt < 2; ++kt) {
        v16h a = load_a_rm(myT, HDIM, lane, kt * 32);
#pragma unroll
        for (int nt = 0; nt < 4; ++nt) cp[nt] = wmma_f16(a, load_b_tr(sWc1, HDIM, lane, kt * 32, nt * 16), cp[nt]);
      }
#pragma unroll
      for (int nt = 0; nt < 4; ++nt) {
        float bia = sB[2][nt * 16 + Nl];
#pragma unroll
        for (int r = 0; r < 8; ++r) cp[nt][r] = silu_f(cp[nt][r] + bia);
        store_c_rm(myT, HDIM, lane, nt * 16, cp[nt]);
      }
      if (lane < 16) {                 // lane = edge index within tile
        float cval = 0.f;
#pragma unroll
        for (int ch = 0; ch < HDIM; ++ch) cval += (float)myT[lane * HDIM + ch] * (float)sWc2[ch];
        float em  = emrow[j0 + lane];
        float coef = cval * em / (sqrtf(radial + 1e-8f) + 1.0f);   // norm_diff
        atomicAdd(&sXacc[ri][0], dx0 * coef);
        atomicAdd(&sXacc[ri][1], dx1 * coef);
        atomicAdd(&sXacc[ri][2], dx2 * coef);
      }
    } // jt

    if (lane < 16) {
#pragma unroll
      for (int nt = 0; nt < 4; ++nt) sNode[ri][HDIM + nt * 16 + lane] = (h16)magg[nt];
    }
  } // rr
  __syncthreads();

  // ---- node MLP for this block's 16 rows (wave 0): [16,128]@[128,64] -> silu -> @[64,64]
  if (wv == 0) {
    v8f o[4] = {zero8(), zero8(), zero8(), zero8()};
#pragma unroll
    for (int kt = 0; kt < 4; ++kt) {
      v16h a = load_a_rm(&sNode[0][0], 128, lane, kt * 32);
#pragma unroll
      for (int nt = 0; nt < 4; ++nt) o[nt] = wmma_f16(a, load_b_tr(sWn1, 128, lane, kt * 32, nt * 16), o[nt]);
    }
#pragma unroll
    for (int nt = 0; nt < 4; ++nt) {
      float bia = sB[3][nt * 16 + Nl];
#pragma unroll
      for (int r = 0; r < 8; ++r) o[nt][r] = silu_f(o[nt][r] + bia);
      store_c_rm(sTmp[0], HDIM, lane, nt * 16, o[nt]);
    }
    v8f p[4] = {zero8(), zero8(), zero8(), zero8()};
#pragma unroll
    for (int kt = 0; kt < 2; ++kt) {
      v16h a = load_a_rm(sTmp[0], HDIM, lane, kt * 32);
#pragma unroll
      for (int nt = 0; nt < 4; ++nt) p[nt] = wmma_f16(a, load_b_tr(sWn2, HDIM, lane, kt * 32, nt * 16), p[nt]);
    }
#pragma unroll
    for (int r = 0; r < 8; ++r) {
      int g = nb + r0 + Mb + r;
      float nm = nmask[g];
#pragma unroll
      for (int nt = 0; nt < 4; ++nt) {
        int ch = nt * 16 + Nl;
        hOut[g * HDIM + ch] = (hIn[g * HDIM + ch] + p[nt][r] + sB[4][ch]) * nm;  // residual + mask
      }
    }
    if (lane < 16) {
      int g = nb + r0 + lane;
      float nm = nmask[g];
      xOut[g * 3 + 0] = (sX[r0 + lane][0] + sXacc[lane][0]) * nm;
      xOut[g * 3 + 1] = (sX[r0 + lane][1] + sXacc[lane][1]) * nm;
      xOut[g * 3 + 2] = (sX[r0 + lane][2] + sXacc[lane][2]) * nm;
    }
  }
}

// ---------------------------------------------------------------------------
// Final: vel = (x - x0)*nm, remove per-batch mean, emit [B, N*3].
// ---------------------------------------------------------------------------
__global__ void egnn_final(const float* __restrict__ xF, const float* __restrict__ x0,
                           const float* __restrict__ nmask, float* __restrict__ out) {
  __shared__ float s[4];
  int b = blockIdx.x, i = threadIdx.x;
  if (i < 4) s[i] = 0.f;
  __syncthreads();
  int g = b * NNODE + i;
  float nm = nmask[g];
  float v0 = (xF[g * 3 + 0] - x0[g * 3 + 0]) * nm;
  float v1 = (xF[g * 3 + 1] - x0[g * 3 + 1]) * nm;
  float v2 = (xF[g * 3 + 2] - x0[g * 3 + 2]) * nm;
  atomicAdd(&s[0], v0); atomicAdd(&s[1], v1); atomicAdd(&s[2], v2); atomicAdd(&s[3], nm);
  __syncthreads();
  float inv = 1.0f / s[3];
  out[g * 3 + 0] = v0 - s[0] * inv * nm;
  out[g * 3 + 1] = v1 - s[1] * inv * nm;
  out[g * 3 + 2] = v2 - s[2] * inv * nm;
}

extern "C" void kernel_launch(void* const* d_in, const int* in_sizes, int n_in,
                              void* d_out, int out_size, void* d_ws, size_t ws_size,
                              hipStream_t stream) {
  (void)in_sizes; (void)n_in; (void)out_size; (void)ws_size;
  const float* t     = (const float*)d_in[0];
  const float* xs    = (const float*)d_in[1];
  const float* hfeat = (const float*)d_in[2];
  const float* nmask = (const float*)d_in[3];
  const float* emask = (const float*)d_in[4];
  // d_in[5], d_in[6] (rows/cols) not needed: edge structure is dense per-batch
  const float* Wemb  = (const float*)d_in[7];
  const float* bemb  = (const float*)d_in[8];
  const float* We1   = (const float*)d_in[9];
  const float* be1   = (const float*)d_in[10];
  const float* We2   = (const float*)d_in[11];
  const float* be2   = (const float*)d_in[12];
  const float* Wc1   = (const float*)d_in[13];
  const float* bc1   = (const float*)d_in[14];
  const float* Wc2   = (const float*)d_in[15];
  const float* Wn1   = (const float*)d_in[16];
  const float* bn1   = (const float*)d_in[17];
  const float* Wn2   = (const float*)d_in[18];
  const float* bn2   = (const float*)d_in[19];

  char* ws = (char*)d_ws;
  float* x0  = (float*)(ws + 0);          // BN*3 f32
  float* xA  = (float*)(ws + 98304);
  float* xB  = (float*)(ws + 196608);
  float* hA  = (float*)(ws + 294912);     // BN*64 f32
  float* hB  = (float*)(ws + 2392064);
  h16* We1h  = (h16*)(ws + 4489216);      // L*64*160 f16
  h16* We2h  = (h16*)(ws + 4571136);      // L*64*64
  h16* Wc1h  = (h16*)(ws + 4603904);
  h16* Wc2h  = (h16*)(ws + 4636672);      // L*64
  h16* Wn1h  = (h16*)(ws + 4637184);      // L*64*128
  h16* Wn2h  = (h16*)(ws + 4702720);      // ends 4735488

  const int wtotal = NLAYER * HDIM * KEDGE + 3 * (NLAYER * HDIM * HDIM) +
                     NLAYER * HDIM + NLAYER * HDIM * 128;
  egnn_prep_weights<<<(wtotal + 255) / 256, 256, 0, stream>>>(
      We1, We2, Wc1, Wc2, Wn1, Wn2, We1h, We2h, Wc1h, Wc2h, Wn1h, Wn2h);
  egnn_prep_nodes<<<(BNODES * HDIM + 255) / 256, 256, 0, stream>>>(
      t, xs, hfeat, nmask, Wemb, bemb, x0, xA, hA);

  float* xb[2] = {xA, xB};
  float* hb[2] = {hA, hB};
  for (int l = 0; l < NLAYER; ++l) {
    egnn_layer<<<dim3(NNODE / 16, NBATCH), 256, 0, stream>>>(
        xb[l & 1], xb[(l + 1) & 1], hb[l & 1], hb[(l + 1) & 1], x0, nmask, emask,
        We1h + l * HDIM * KEDGE, We2h + l * HDIM * HDIM,
        Wc1h + l * HDIM * HDIM, Wc2h + l * HDIM,
        Wn1h + l * HDIM * 128, Wn2h + l * HDIM * HDIM,
        be1 + l * HDIM, be2 + l * HDIM, bc1 + l * HDIM,
        bn1 + l * HDIM, bn2 + l * HDIM);
  }
  egnn_final<<<NBATCH, NNODE, 0, stream>>>(xb[NLAYER & 1], x0, nmask, (float*)d_out);
}
